// MetaBlock_82377472737839
// MI455X (gfx1250) — matrix operand order repaired and checked by reading, not compiled, not run on hardware
//
#include <hip/hip_runtime.h>

// ---------------------------------------------------------------------------
// MI455X (gfx1250) AFNO block.
//  * All GEMM work (proj, DFT/IDFT as 64-pt DFT-matrix GEMMs, w1/w2/w3, MLP)
//    runs on v_wmma_f32_16x16x32_bf16 (bf16 in, f32 accumulate).
//  * 128x128x32 double-buffered tiles for the large GEMMs (8 WMMA/wave/step),
//    64x64x32 for the M=64 DFT batches. float4 staging loads where a unit
//    stride exists (always, in this workload).
//  * softmax over nb==1 is identically 1 -> cross/sim path eliminated.
// ---------------------------------------------------------------------------

typedef __attribute__((ext_vector_type(16))) __bf16 v16bf;
typedef __attribute__((ext_vector_type(8)))  float  v8f;

union Frag16 { unsigned int u[8]; v16bf v; };

__device__ __forceinline__ unsigned short f2bf(float f) {
  union { float f; unsigned int u; } c; c.f = f;
  unsigned int u = c.u;
  u += 0x7FFFu + ((u >> 16) & 1u);          // round-to-nearest-even
  return (unsigned short)(u >> 16);
}

#define TK 32
#define LPAD 36   // LDS row pitch in bf16 elems (72B, 4B aligned, bank-skewed)

// Stage a R x 32 tile G[(r0+r)*rs + (k0+k)*cs] -> regs (R*32/256 per thread).
// mode 0: cs==1 (float4 along k), mode 1: rs==1 (float4 along r), 2: scalar.
template <int R>
__device__ __forceinline__ void tile_load(float* regs, const float* __restrict__ G,
                                          long rs, long cs, int r0, int k0, int mode) {
  const int t = threadIdx.x;
  if (mode == 0) {
#pragma unroll
    for (int i = 0; i < R / 32; ++i) {
      int s = t + i * 256;
      int r = s >> 3, k4 = (s & 7) << 2;
      const float4 q = *(const float4*)(G + (long)(r0 + r) * rs + (k0 + k4));
      regs[4*i+0] = q.x; regs[4*i+1] = q.y; regs[4*i+2] = q.z; regs[4*i+3] = q.w;
    }
  } else if (mode == 1) {
    constexpr int RQ = R / 4;
#pragma unroll
    for (int i = 0; i < R / 32; ++i) {
      int s = t + i * 256;
      int k = s / RQ, r4 = (s % RQ) << 2;
      const float4 q = *(const float4*)(G + (long)(r0 + r4) + (long)(k0 + k) * cs);
      regs[4*i+0] = q.x; regs[4*i+1] = q.y; regs[4*i+2] = q.z; regs[4*i+3] = q.w;
    }
  } else {
#pragma unroll
    for (int i = 0; i < R / 8; ++i) {
      int s = t + i * 256;
      int r = s >> 5, kk = s & 31;
      regs[i] = G[(long)(r0 + r) * rs + (long)(k0 + kk) * cs];
    }
  }
}

template <int R>
__device__ __forceinline__ void tile_store(unsigned short* S, const float* regs, int mode) {
  const int t = threadIdx.x;
  if (mode == 0) {
#pragma unroll
    for (int i = 0; i < R / 32; ++i) {
      int s = t + i * 256;
      int r = s >> 3, k4 = (s & 7) << 2;
      unsigned int p0 = (unsigned)f2bf(regs[4*i+0]) | ((unsigned)f2bf(regs[4*i+1]) << 16);
      unsigned int p1 = (unsigned)f2bf(regs[4*i+2]) | ((unsigned)f2bf(regs[4*i+3]) << 16);
      unsigned int* d = (unsigned int*)(S + r * LPAD + k4);
      d[0] = p0; d[1] = p1;
    }
  } else if (mode == 1) {
    constexpr int RQ = R / 4;
#pragma unroll
    for (int i = 0; i < R / 32; ++i) {
      int s = t + i * 256;
      int k = s / RQ, r4 = (s % RQ) << 2;
#pragma unroll
      for (int j = 0; j < 4; ++j) S[(r4 + j) * LPAD + k] = f2bf(regs[4*i+j]);
    }
  } else {
#pragma unroll
    for (int i = 0; i < R / 8; ++i) {
      int s = t + i * 256;
      int r = s >> 5, kk = s & 31;
      S[r * LPAD + kk] = f2bf(regs[i]);
    }
  }
}

// C[m,n] = (accf?C:0) + alpha*sum_k A[m,k]*B[k,n] (+bias[n]);
// ep 1 -> exact GELU, ep 2 -> softshrink(0.01).
// Strided A/B/C, batch z = z0*nb1 + z1. Wave computes WMR x WNR 16x16 tiles.
template <int BM, int BN, int WMR, int WNR>
__global__ __launch_bounds__(256) void gemm_bf16_wmma(
    const float* __restrict__ A, long ars, long acs, long abs0, long abs1,
    const float* __restrict__ Bm, long bks, long bns, long bbs0, long bbs1,
    float* __restrict__ C, long crs, long ccs, long cbs0, long cbs1,
    const float* __restrict__ bias,
    int K, int nb1, float alpha, int accf, int ep)
{
  constexpr int WAVES_M = BM / (16 * WMR);
  constexpr int WAVES_N = BN / (16 * WNR);
  static_assert(WAVES_M * WAVES_N == 8, "8 waves");

  __shared__ unsigned short As[2][BM * LPAD];
  __shared__ unsigned short Bs[2][BN * LPAD];

  const int z  = blockIdx.z;
  const int z0 = z / nb1, z1 = z - z0 * nb1;
  A  += (long)z0 * abs0 + (long)z1 * abs1;
  Bm += (long)z0 * bbs0 + (long)z1 * bbs1;
  C  += (long)z0 * cbs0 + (long)z1 * cbs1;

  const int m0 = blockIdx.y * BM;
  const int n0 = blockIdx.x * BN;
  const int t = threadIdx.x;
  const int wave = t >> 5, lane = t & 31;
  const int wm = wave % WAVES_M, wn = wave / WAVES_M;
  const int half = lane >> 4, r16 = lane & 15;

  const int amode = (acs == 1) ? 0 : ((ars == 1) ? 1 : 2);
  const int bmode = (bks == 1) ? 0 : ((bns == 1) ? 1 : 2);

  v8f zero = {};
  v8f acc[WMR][WNR];
#pragma unroll
  for (int im = 0; im < WMR; ++im)
#pragma unroll
    for (int in_ = 0; in_ < WNR; ++in_) acc[im][in_] = zero;

  float ra[BM / 8], rb[BN / 8];
  tile_load<BM>(ra, A, ars, acs, m0, 0, amode);
  tile_load<BN>(rb, Bm, bns, bks, n0, 0, bmode);
  tile_store<BM>(As[0], ra, amode);
  tile_store<BN>(Bs[0], rb, bmode);

  const int nk = K / TK;
  for (int i = 0; i < nk; ++i) {
    __syncthreads();
    const int buf = i & 1;
    if (i + 1 < nk) {                       // prefetch next tile into registers
      tile_load<BM>(ra, A, ars, acs, m0, (i + 1) * TK, amode);
      tile_load<BN>(rb, Bm, bns, bks, n0, (i + 1) * TK, bmode);
    }
    // A frag (16x32 bf16): lane<16 holds K {0..7,16..23}; lane>=16 offset +8.
    Frag16 af[WMR];
#pragma unroll
    for (int im = 0; im < WMR; ++im) {
      const unsigned short* ap =
          &As[buf][((wm * WMR + im) * 16 + r16) * LPAD + 8 * half];
#pragma unroll
      for (int j = 0; j < 4; ++j) {
        af[im].u[j]     = *(const unsigned int*)(ap + 2 * j);
        af[im].u[j + 4] = *(const unsigned int*)(ap + 16 + 2 * j);
      }
    }
    // B frag (32x16 bf16): lane<16 holds K {0..7,16..23}; lane>=16 offset +8.
#pragma unroll
    for (int in_ = 0; in_ < WNR; ++in_) {
      Frag16 bf_;
      const unsigned short* bp =
          &Bs[buf][((wn * WNR + in_) * 16 + r16) * LPAD + 8 * half];
#pragma unroll
      for (int j = 0; j < 4; ++j) {
        bf_.u[j]     = *(const unsigned int*)(bp + 2 * j);
        bf_.u[j + 4] = *(const unsigned int*)(bp + 16 + 2 * j);
      }
#pragma unroll
      for (int im = 0; im < WMR; ++im)
        acc[im][in_] = __builtin_amdgcn_wmma_f32_16x16x32_bf16(
            false, af[im].v, false, bf_.v, (short)0, acc[im][in_], false, false);
    }
    if (i + 1 < nk) {
      tile_store<BM>(As[buf ^ 1], ra, amode);
      tile_store<BN>(Bs[buf ^ 1], rb, bmode);
    }
  }

  // Epilogue. C layout: VGPR r -> M = r + 8*half, N = lane&15.
  float oldv[WMR][WNR][8];
  if (accf) {
#pragma unroll
    for (int im = 0; im < WMR; ++im)
#pragma unroll
      for (int in_ = 0; in_ < WNR; ++in_)
#pragma unroll
        for (int r = 0; r < 8; ++r) {
          const int mm = m0 + (wm * WMR + im) * 16 + half * 8 + r;
          const int nn = n0 + (wn * WNR + in_) * 16 + r16;
          oldv[im][in_][r] = C[(long)mm * crs + (long)nn * ccs];
        }
  }
#pragma unroll
  for (int im = 0; im < WMR; ++im)
#pragma unroll
    for (int in_ = 0; in_ < WNR; ++in_)
#pragma unroll
      for (int r = 0; r < 8; ++r) {
        const int mm = m0 + (wm * WMR + im) * 16 + half * 8 + r;
        const int nn = n0 + (wn * WNR + in_) * 16 + r16;
        const long idx = (long)mm * crs + (long)nn * ccs;
        float v = alpha * acc[im][in_][r];
        if (bias) v += bias[nn];
        if (accf) v += oldv[im][in_][r];
        if (ep == 1)      v = 0.5f * v * (1.0f + erff(v * 0.70710678118654752f));
        else if (ep == 2) v = (v > 0.01f) ? (v - 0.01f)
                            : ((v < -0.01f) ? (v + 0.01f) : 0.0f);
        C[idx] = v;
      }
}

// ---------------------------------------------------------------------------
// LayerNorm over C channels; one wave per token (wave32), NV = C/32 in regs.
template <int NV>
__global__ __launch_bounds__(256) void ln_kernel(
    const float* __restrict__ in, const float* __restrict__ g,
    const float* __restrict__ bt, float* __restrict__ out,
    long bis, long nis, long cs, int ntok_per_b)
{
  const int C = NV * 32;
  const int wave = threadIdx.x >> 5, lane = threadIdx.x & 31;
  const long tok = (long)blockIdx.x * 8 + wave;
  const int b = (int)(tok / ntok_per_b);
  const int n = (int)(tok - (long)b * ntok_per_b);
  const float* base = in + (long)b * bis + (long)n * nis;

  float v[NV];
  float s = 0.f;
#pragma unroll
  for (int i = 0; i < NV; ++i) {
    v[i] = base[(long)(lane + (i << 5)) * cs];
    s += v[i];
  }
#pragma unroll
  for (int m = 16; m >= 1; m >>= 1) s += __shfl_xor(s, m, 32);
  const float mu = s / (float)C;
  float q = 0.f;
#pragma unroll
  for (int i = 0; i < NV; ++i) { float d = v[i] - mu; q += d * d; }
#pragma unroll
  for (int m = 16; m >= 1; m >>= 1) q += __shfl_xor(q, m, 32);
  const float rs = rsqrtf(q / (float)C + 1e-5f);
  float* ob = out + tok * C;
#pragma unroll
  for (int i = 0; i < NV; ++i) {
    int ch = lane + (i << 5);
    ob[ch] = (v[i] - mu) * rs * g[ch] + bt[ch];
  }
}

// ---------------------------------------------------------------------------
// 64-pt DFT matrix (ortho 1/8 per axis): D = cos(t) - i sin(t).
__global__ void dft_kernel(float* __restrict__ Dr, float* __restrict__ Di) {
  int i = blockIdx.x * 256 + threadIdx.x;       // 4096 entries
  int j = i >> 6, k = i & 63;
  float t = (float)((j * k) & 63) * 0.0981747704246810387f;  // 2*pi/64
  Dr[i] = cosf(t) * 0.125f;
  Di[i] = -sinf(t) * 0.125f;
}

// Pangu-guided frequency modulation (softmax over nb==1 == 1 -> mag = |x|).
__global__ void mod_kernel(float* __restrict__ Xr, float* __restrict__ Xi,
                           const float* __restrict__ Pr, const float* __restrict__ Pi,
                           const float* __restrict__ alpha, long n)
{
  long i = (long)blockIdx.x * 256 + threadIdx.x;
  if (i >= n) return;
  const int c = (int)(i & 255);
  float xr = Xr[i], xi = Xi[i], pr = Pr[i], pi = Pi[i];
  float a  = alpha[c];
  float mo = sqrtf(xr * xr + xi * xi);
  float mp = sqrtf(pr * pr + pi * pi);
  float cx = mo > 0.f ? xr / mo : 1.f;
  float sx = mo > 0.f ? xi / mo : 0.f;
  float cp = mp > 0.f ? pr / mp : 1.f;
  float sp = mp > 0.f ? pi / mp : 0.f;
  float sc = a * cx + (1.f - a) * cp;
  float ss = a * sx + (1.f - a) * sp;
  float rn = rsqrtf(sc * sc + ss * ss + 1e-8f);
  Xr[i] = fmaxf(mo * sc * rn, 0.f);
  Xi[i] = fmaxf(mo * ss * rn, 0.f);
}

__global__ void add_kernel(float* __restrict__ h, const float* __restrict__ f, long n) {
  long i = (long)blockIdx.x * 256 + threadIdx.x;
  if (i < n) h[i] += f[i];
}

// out[b,c,h,w] = out2[b,n,c] + h[b,n,c]
__global__ void final_kernel(const float* __restrict__ out2,
                             const float* __restrict__ h,
                             float* __restrict__ out, long n)
{
  long i = (long)blockIdx.x * 256 + threadIdx.x;
  if (i >= n) return;
  const int  c  = (int)(i & 255);
  const long bn = i >> 8;            // b*4096 + tok
  const long b  = bn >> 12;
  const long nn = bn & 4095;
  out[(b << 20) + ((long)c << 12) + nn] = out2[i] + h[i];
}

// ---------------------------------------------------------------------------
static void gemm(hipStream_t s,
                 const float* A, long ars, long acs, long abs0, long abs1,
                 const float* B, long bks, long bns, long bbs0, long bbs1,
                 float* C, long crs, long ccs, long cbs0, long cbs1,
                 const float* bias, int M, int N, int K, int nbz, int nb1,
                 float alpha, int acc, int ep)
{
  if ((M % 128 == 0) && (N % 128 == 0)) {
    dim3 g(N / 128, M / 128, nbz);
    gemm_bf16_wmma<128, 128, 2, 4><<<g, 256, 0, s>>>(
        A, ars, acs, abs0, abs1, B, bks, bns, bbs0, bbs1,
        C, crs, ccs, cbs0, cbs1, bias, K, nb1, alpha, acc, ep);
  } else {
    dim3 g(N / 64, M / 64, nbz);
    gemm_bf16_wmma<64, 64, 1, 2><<<g, 256, 0, s>>>(
        A, ars, acs, abs0, abs1, B, bks, bns, bbs0, bbs1,
        C, crs, ccs, cbs0, cbs1, bias, K, nb1, alpha, acc, ep);
  }
}

extern "C" void kernel_launch(void* const* d_in, const int* in_sizes, int n_in,
                              void* d_out, int out_size, void* d_ws, size_t ws_size,
                              hipStream_t stream) {
  (void)in_sizes; (void)n_in; (void)out_size; (void)ws_size;
  const float* x      = (const float*)d_in[0];   // (8,128,64,64)
  const float* xp     = (const float*)d_in[1];   // (8,256,64,64)
  const float* w1     = (const float*)d_in[3];   // (2,1,128,256)
  const float* b1     = (const float*)d_in[4];   // (2,1,256)
  const float* w2     = (const float*)d_in[5];   // (2,1,256,256)
  const float* b2     = (const float*)d_in[6];
  const float* w3     = (const float*)d_in[7];
  const float* b3     = (const float*)d_in[8];
  const float* alpha  = (const float*)d_in[9];   // 256
  const float* gamma1 = (const float*)d_in[10];
  const float* beta1  = (const float*)d_in[11];
  const float* gamma2 = (const float*)d_in[12];
  const float* beta2  = (const float*)d_in[13];
  const float* projw  = (const float*)d_in[14];  // (256,128)
  const float* projb  = (const float*)d_in[15];
  const float* fc1w   = (const float*)d_in[16];  // (2048,256)
  const float* fc1b   = (const float*)d_in[17];
  const float* fc2w   = (const float*)d_in[18];  // (256,2048)
  const float* fc2b   = (const float*)d_in[19];
  float* out = (float*)d_out;

  const long S128 = 8L * 4096 * 128;
  const long S256 = 8L * 4096 * 256;

  float* W   = (float*)d_ws;
  float* Dr  = W;            float* Di  = Dr + 4096;
  float* xn  = Di + 4096;                      // (b,n,128)
  float* t1r = xn  + S128;   float* t1i = t1r + S128;
  float* Xr0 = t1i + S128;   float* Xi0 = Xr0 + S128;   // fft(xn)
  float* Xr  = Xi0 + S128;   float* Xi  = Xr  + S256;   // w1 out / modulated
  float* p1r = Xi  + S256;   float* p1i = p1r + S256;
  float* PFr = p1i + S256;   float* PFi = PFr + S256;   // fft(xp)
  float* Pr  = PFi + S256;   float* Pi  = Pr  + S256;   // w3 out
  float* res = Pi  + S256;                               // residual -> h
  // reuse (lifetimes verified): O2 <- p1, U <- PF, f <- Pr, m <- Xr,
  // mid <- Xi (per-batch 4096x2048), out2 <- Pi.
  float* O2r = p1r; float* O2i = p1i;
  float* Ur  = PFr; float* Ui  = PFi;
  float* fbuf = Pr; float* m = Xr; float* mid = Xi; float* out2 = Pi;

  dft_kernel<<<16, 256, 0, stream>>>(Dr, Di);

  // (1) residual = tok @ proj_w.T + proj_b ; tok[m,k] = x[b,k,m]
  gemm(stream, x, 1, 4096, 524288, 0,
       projw, 1, 128, 0, 0,
       res, 256, 1, 1048576, 0,
       projb, 4096, 256, 128, 8, 1, 1.f, 0, 0);

  // (2) xn = LN(tok) over 128 ch
  ln_kernel<4><<<4096, 256, 0, stream>>>(x, gamma1, beta1, xn, 524288, 1, 4096, 4096);

  // (3) DFT along W on xn (real input): t1 = D @ xn  per (b,h)
  gemm(stream, Dr, 64, 1, 0, 0, xn, 128, 1, 524288, 8192,
       t1r, 128, 1, 524288, 8192, nullptr, 64, 128, 64, 512, 64, 1.f, 0, 0);
  gemm(stream, Di, 64, 1, 0, 0, xn, 128, 1, 524288, 8192,
       t1i, 128, 1, 524288, 8192, nullptr, 64, 128, 64, 512, 64, 1.f, 0, 0);

  // (4) DFT along H: X = D @ t1 (complex) per (b,w)
  gemm(stream, Dr, 64, 1, 0, 0, t1r, 8192, 1, 524288, 128,
       Xr0, 8192, 1, 524288, 128, nullptr, 64, 128, 64, 512, 64, 1.f, 0, 0);
  gemm(stream, Di, 64, 1, 0, 0, t1i, 8192, 1, 524288, 128,
       Xr0, 8192, 1, 524288, 128, nullptr, 64, 128, 64, 512, 64, -1.f, 1, 0);
  gemm(stream, Dr, 64, 1, 0, 0, t1i, 8192, 1, 524288, 128,
       Xi0, 8192, 1, 524288, 128, nullptr, 64, 128, 64, 512, 64, 1.f, 0, 0);
  gemm(stream, Di, 64, 1, 0, 0, t1r, 8192, 1, 524288, 128,
       Xi0, 8192, 1, 524288, 128, nullptr, 64, 128, 64, 512, 64, 1.f, 1, 0);

  // (5) w1 complex channel GEMM (128->256): X = Xf @ W1 + b1
  const float* W1r = w1;            const float* W1i = w1 + 32768;
  gemm(stream, Xr0, 128, 1, 524288, 0, W1r, 256, 1, 0, 0,
       Xr, 256, 1, 1048576, 0, b1,       4096, 256, 128, 8, 1, 1.f, 0, 0);
  gemm(stream, Xi0, 128, 1, 524288, 0, W1i, 256, 1, 0, 0,
       Xr, 256, 1, 1048576, 0, nullptr,  4096, 256, 128, 8, 1, -1.f, 1, 0);
  gemm(stream, Xi0, 128, 1, 524288, 0, W1r, 256, 1, 0, 0,
       Xi, 256, 1, 1048576, 0, b1 + 256, 4096, 256, 128, 8, 1, 1.f, 0, 0);
  gemm(stream, Xr0, 128, 1, 524288, 0, W1i, 256, 1, 0, 0,
       Xi, 256, 1, 1048576, 0, nullptr,  4096, 256, 128, 8, 1, 1.f, 1, 0);

  // (6) DFT along W on x_pangu (read NCHW directly via strides)
  gemm(stream, Dr, 64, 1, 0, 0, xp, 1, 4096, 1048576, 64,
       p1r, 256, 1, 1048576, 16384, nullptr, 64, 256, 64, 512, 64, 1.f, 0, 0);
  gemm(stream, Di, 64, 1, 0, 0, xp, 1, 4096, 1048576, 64,
       p1i, 256, 1, 1048576, 16384, nullptr, 64, 256, 64, 512, 64, 1.f, 0, 0);

  // (7) DFT along H on pangu
  gemm(stream, Dr, 64, 1, 0, 0, p1r, 16384, 1, 1048576, 256,
       PFr, 16384, 1, 1048576, 256, nullptr, 64, 256, 64, 512, 64, 1.f, 0, 0);
  gemm(stream, Di, 64, 1, 0, 0, p1i, 16384, 1, 1048576, 256,
       PFr, 16384, 1, 1048576, 256, nullptr, 64, 256, 64, 512, 64, -1.f, 1, 0);
  gemm(stream, Dr, 64, 1, 0, 0, p1i, 16384, 1, 1048576, 256,
       PFi, 16384, 1, 1048576, 256, nullptr, 64, 256, 64, 512, 64, 1.f, 0, 0);
  gemm(stream, Di, 64, 1, 0, 0, p1r, 16384, 1, 1048576, 256,
       PFi, 16384, 1, 1048576, 256, nullptr, 64, 256, 64, 512, 64, 1.f, 1, 0);

  // (8) w3 complex GEMM (256->256): P = PF @ W3 + b3
  const float* W3r = w3;            const float* W3i = w3 + 65536;
  gemm(stream, PFr, 256, 1, 1048576, 0, W3r, 256, 1, 0, 0,
       Pr, 256, 1, 1048576, 0, b3,       4096, 256, 256, 8, 1, 1.f, 0, 0);
  gemm(stream, PFi, 256, 1, 1048576, 0, W3i, 256, 1, 0, 0,
       Pr, 256, 1, 1048576, 0, nullptr,  4096, 256, 256, 8, 1, -1.f, 1, 0);
  gemm(stream, PFi, 256, 1, 1048576, 0, W3r, 256, 1, 0, 0,
       Pi, 256, 1, 1048576, 0, b3 + 256, 4096, 256, 256, 8, 1, 1.f, 0, 0);
  gemm(stream, PFr, 256, 1, 1048576, 0, W3i, 256, 1, 0, 0,
       Pi, 256, 1, 1048576, 0, nullptr,  4096, 256, 256, 8, 1, 1.f, 1, 0);

  // (9) phase modulation (in-place into Xr/Xi)
  mod_kernel<<<(unsigned)(S256 / 256), 256, 0, stream>>>(Xr, Xi, Pr, Pi, alpha, S256);

  // (10) w2 complex GEMM + softshrink epilogue on the accumulating pass
  const float* W2r = w2;            const float* W2i = w2 + 65536;
  gemm(stream, Xr, 256, 1, 1048576, 0, W2r, 256, 1, 0, 0,
       O2r, 256, 1, 1048576, 0, b2,       4096, 256, 256, 8, 1, 1.f, 0, 0);
  gemm(stream, Xi, 256, 1, 1048576, 0, W2i, 256, 1, 0, 0,
       O2r, 256, 1, 1048576, 0, nullptr,  4096, 256, 256, 8, 1, -1.f, 1, 2);
  gemm(stream, Xi, 256, 1, 1048576, 0, W2r, 256, 1, 0, 0,
       O2i, 256, 1, 1048576, 0, b2 + 256, 4096, 256, 256, 8, 1, 1.f, 0, 0);
  gemm(stream, Xr, 256, 1, 1048576, 0, W2i, 256, 1, 0, 0,
       O2i, 256, 1, 1048576, 0, nullptr,  4096, 256, 256, 8, 1, 1.f, 1, 2);

  // (11) IDFT along H: U = conj(D) @ O2
  gemm(stream, Dr, 64, 1, 0, 0, O2r, 16384, 1, 1048576, 256,
       Ur, 16384, 1, 1048576, 256, nullptr, 64, 256, 64, 512, 64, 1.f, 0, 0);
  gemm(stream, Di, 64, 1, 0, 0, O2i, 16384, 1, 1048576, 256,
       Ur, 16384, 1, 1048576, 256, nullptr, 64, 256, 64, 512, 64, 1.f, 1, 0);
  gemm(stream, Dr, 64, 1, 0, 0, O2i, 16384, 1, 1048576, 256,
       Ui, 16384, 1, 1048576, 256, nullptr, 64, 256, 64, 512, 64, 1.f, 0, 0);
  gemm(stream, Di, 64, 1, 0, 0, O2r, 16384, 1, 1048576, 256,
       Ui, 16384, 1, 1048576, 256, nullptr, 64, 256, 64, 512, 64, -1.f, 1, 0);

  // (12) IDFT along W, real part only: f = Dr@Ur + Di@Ui
  gemm(stream, Dr, 64, 1, 0, 0, Ur, 256, 1, 1048576, 16384,
       fbuf, 256, 1, 1048576, 16384, nullptr, 64, 256, 64, 512, 64, 1.f, 0, 0);
  gemm(stream, Di, 64, 1, 0, 0, Ui, 256, 1, 1048576, 16384,
       fbuf, 256, 1, 1048576, 16384, nullptr, 64, 256, 64, 512, 64, 1.f, 1, 0);

  // (13) h = f + residual (in res buffer)
  add_kernel<<<(unsigned)(S256 / 256), 256, 0, stream>>>(res, fbuf, S256);

  // (14) m = LN(h) over 256 ch
  ln_kernel<8><<<4096, 256, 0, stream>>>(res, gamma2, beta2, m, 1048576, 256, 1, 4096);

  // (15) MLP per batch (bounded scratch): mid = gelu(m@fc1.T+b); out2 = mid@fc2.T+b
  for (int b = 0; b < 8; ++b) {
    gemm(stream, m + (long)b * 1048576, 256, 1, 0, 0,
         fc1w, 1, 256, 0, 0,
         mid, 2048, 1, 0, 0,
         fc1b, 4096, 2048, 256, 1, 1, 1.f, 0, 1);
    gemm(stream, mid, 2048, 1, 0, 0,
         fc2w, 1, 2048, 0, 0,
         out2 + (long)b * 1048576, 256, 1, 0, 0,
         fc2b, 4096, 256, 2048, 1, 1, 1.f, 0, 0);
  }

  // (16) out[b,c,h,w] = out2 + h (transposed store)
  final_kernel<<<(unsigned)(S256 / 256), 256, 0, stream>>>(out2, res, out, S256);
}
